// WindowAttention2D_79070347919713
// MI455X (gfx1250) — matrix operand order repaired and checked by reading, not compiled, not run on hardware
//
#include <hip/hip_runtime.h>

// Fused window attention for gfx1250 (CDNA5, wave32, WMMA f16 16x16x32).
// One workgroup (256 threads = 8 waves) per window of 49 tokens, dim 192,
// 6 heads x head_dim 32. All intermediates live in LDS. Weights are
// pre-converted to f16 in d_ws by a tiny prep kernel (q rows pre-scaled),
// so the GEMM inner loops are pure b128 loads + v_wmma.

typedef __attribute__((ext_vector_type(16))) _Float16 v16h;
typedef __attribute__((ext_vector_type(8)))  _Float16 v8h;
typedef __attribute__((ext_vector_type(4)))  _Float16 v4h;
typedef __attribute__((ext_vector_type(8)))  float    v8f;
typedef __attribute__((ext_vector_type(4)))  float    v4f;

#define SEQ 49
#define NP  64
#define DIM 192
#define NH  6
#define HD  32
#define QKV_ELEMS (3 * DIM * DIM)   // 110592
#define PRJ_ELEMS (DIM * DIM)       // 36864
#define WS_BYTES  ((QKV_ELEMS + PRJ_ELEMS) * 2)
#define VOFF (2 * NH * NP * HD)     // offset of transposed v inside qkv_s

static __device__ const float kQScale = 0.17677669529663687f;  // 32^-0.5

// A-matrix fragment (16x32 f16, MxK), row-major source.
// lane m = lane&15; half = lane>>4:
//   slot e -> K = kbase + half*16 + e (16 contiguous f16)
__device__ __forceinline__ v16h frag_a_lds(const _Float16* rowptr, int kbase, int half) {
  const v8h* p0 = (const v8h*)(rowptr + kbase + half * 16);
  const v8h* p1 = (const v8h*)(rowptr + kbase + half * 16 + 8);
  v8h c0 = *p0, c1 = *p1;
  v16h r;
#pragma unroll
  for (int i = 0; i < 8; ++i) { r[i] = c0[i]; r[i + 8] = c1[i]; }
  return r;
}

// B-matrix fragment (32x16 f16, KxN); lane n = lane&15, half = lane>>4,
// slot e -> K = kbase + half*16 + e (16 contiguous f16).
__device__ __forceinline__ v16h frag_b_f16(const _Float16* rowptr, int kbase, int half) {
  const v8h* p0 = (const v8h*)(rowptr + kbase + half * 16);
  const v8h* p1 = (const v8h*)(rowptr + kbase + half * 16 + 8);
  v8h c0 = *p0, c1 = *p1;
  v16h r;
#pragma unroll
  for (int i = 0; i < 8; ++i) { r[i] = c0[i]; r[i + 8] = c1[i]; }
  return r;
}

// Fallback: B fragment from global f32 weights, convert to f16 on the fly.
__device__ __forceinline__ v16h frag_b_gw(const float* __restrict__ W, int nrow,
                                          int ld, int kbase, int half) {
  const float* p = W + (size_t)nrow * ld + kbase + half * 16;
  v16h r;
#pragma unroll
  for (int i = 0; i < 16; ++i) r[i] = (_Float16)p[i];
  return r;
}

// One-shot weight conversion: qkv_w (q rows pre-scaled) and proj_w -> f16.
__global__ __launch_bounds__(256) void convert_weights_kernel(
    const float* __restrict__ qkv_w, const float* __restrict__ proj_w,
    _Float16* __restrict__ wq, _Float16* __restrict__ wp) {
  int i = blockIdx.x * 256 + threadIdx.x;
  if (i < QKV_ELEMS) {
    float v = qkv_w[i];
    if (i < DIM * DIM) v *= kQScale;  // q weight rows
    wq[i] = (_Float16)v;
  }
  if (i < PRJ_ELEMS) wp[i] = (_Float16)proj_w[i];
}

template <bool F16W>
__global__ __launch_bounds__(256) void win_attn_kernel(
    const float* __restrict__ x, const float* __restrict__ qkv_w,
    const float* __restrict__ qkv_b, const float* __restrict__ proj_w,
    const float* __restrict__ proj_b, const _Float16* __restrict__ wq_h,
    const _Float16* __restrict__ wp_h, float* __restrict__ out) {
  // LDS (144 KB):
  //  [0, 49152)      : scores/probs f16 6x64x64; first 24576 B double as x f16 64x192
  //  [49152, 73728)  : attention output f16 64x192
  //  [73728, 147456) : qkv f16: q,k as [t*6+h][64][32]; v transposed [h][32][64]
  __shared__ __align__(16) unsigned char smem[49152 + 24576 + 73728];
  _Float16* x_s   = (_Float16*)smem;
  _Float16* sp_s  = (_Float16*)smem;
  _Float16* ao_s  = (_Float16*)(smem + 49152);
  _Float16* qkv_s = (_Float16*)(smem + 49152 + 24576);

  const int tid  = threadIdx.x;
  const int wave = tid >> 5;
  const int lane = tid & 31;
  const int lrow = lane & 15;
  const int half = lane >> 4;
  const size_t win = blockIdx.x;
  const float* xw = x + win * (SEQ * DIM);

  // ---- Stage 0: stage x into LDS as f16 (vectorized), zero-pad rows 49..63 ----
  {
    const v4f* xv = (const v4f*)xw;               // 9408 floats = 2352 x float4
    for (int i = tid; i < (SEQ * DIM) / 4; i += 256) {
      v4f v = xv[i];
      v4h hv;
#pragma unroll
      for (int j = 0; j < 4; ++j) hv[j] = (_Float16)v[j];
      *(v4h*)(x_s + i * 4) = hv;
    }
    for (int i = tid; i < ((NP - SEQ) * DIM) / 8; i += 256) {
      v8h z = {};
      *(v8h*)(x_s + SEQ * DIM + i * 8) = z;
    }
  }
  __syncthreads();

  // ---- Stage 1: qkv = x @ qkv_w^T + b; scale q; scatter f16 to qkv_s ----
  for (int mt = 0; mt < 4; ++mt) {
    v16h a[6];
#pragma unroll
    for (int kk = 0; kk < 6; ++kk)
      a[kk] = frag_a_lds(x_s + (mt * 16 + lrow) * DIM, kk * 32, half);
    for (int nt = wave; nt < 36; nt += 8) {
      v8f acc = {};
#pragma unroll
      for (int kk = 0; kk < 6; ++kk) {
        v16h b = F16W ? frag_b_f16(wq_h + (size_t)(nt * 16 + lrow) * DIM, kk * 32, half)
                      : frag_b_gw(qkv_w, nt * 16 + lrow, DIM, kk * 32, half);
        acc = __builtin_amdgcn_wmma_f32_16x16x32_f16(false, a[kk], false, b,
                                                     (short)0, acc, false, false);
      }
      // Tile-uniform decode: a 16-wide tile never straddles q/k/v or heads.
      int t   = nt / 12;               // 0=q 1=k 2=v (wave-uniform)
      int sub = nt - t * 12;
      int h   = sub >> 1;              // head (wave-uniform)
      int d   = ((sub & 1) << 4) + lrow;
      float bias = qkv_b[nt * 16 + lrow];
      int row0 = mt * 16 + half * 8;
      if (t == 2) {
        // v stored transposed [h][d][row]: 8 consecutive rows -> one b128 store.
        v8h pk;
#pragma unroll
        for (int r = 0; r < 8; ++r) pk[r] = (_Float16)(acc[r] + bias);
        *(v8h*)(qkv_s + VOFF + (h * HD + d) * NP + row0) = pk;
      } else {
        float mul = (t == 0) ? kQScale : 1.f;
        _Float16* dst = qkv_s + ((t * NH + h) * NP + row0) * HD + d;
#pragma unroll
        for (int r = 0; r < 8; ++r) {
          // f16 path: q weights pre-scaled -> scale bias only; f32: scale both.
          float v = F16W ? (acc[r] + bias * mul) : ((acc[r] + bias) * mul);
          dst[r * HD] = (_Float16)v;
        }
      }
    }
  }
  __syncthreads();

  // ---- Stage 2: per-head S = q @ k^T; 24 (head,mt) pairs over 8 waves ----
  for (int pair = wave; pair < NH * 4; pair += 8) {
    int h = pair >> 2, mt = pair & 3;
    const _Float16* qb = qkv_s + (0 * NH + h) * NP * HD;
    const _Float16* kb = qkv_s + (1 * NH + h) * NP * HD;
    _Float16* sb = sp_s + h * NP * NP;
    v16h a = frag_a_lds(qb + (mt * 16 + lrow) * HD, 0, half);
#pragma unroll
    for (int nt = 0; nt < 4; ++nt) {
      v16h b = frag_b_f16(kb + (nt * 16 + lrow) * HD, 0, half);
      v8f acc = {};
      acc = __builtin_amdgcn_wmma_f32_16x16x32_f16(false, a, false, b,
                                                   (short)0, acc, false, false);
#pragma unroll
      for (int r = 0; r < 8; ++r)
        sb[(mt * 16 + half * 8 + r) * NP + nt * 16 + lrow] = (_Float16)acc[r];
    }
  }
  __syncthreads();

  // ---- Softmax over key dim, in place in LDS (no private arrays) ----
  for (int idx = tid; idx < NH * SEQ; idx += 256) {
    int h = idx / SEQ, row = idx - h * SEQ;
    _Float16* p = sp_s + h * NP * NP + row * NP;
    float m = -1e30f;
    for (int j = 0; j < SEQ; ++j) m = fmaxf(m, (float)p[j]);
    float s = 0.f;
    for (int j = 0; j < SEQ; ++j) {
      float e = __expf((float)p[j] - m);
      s += e;
      p[j] = (_Float16)e;  // unnormalized, rescaled below
    }
    float inv = 1.f / s;
    for (int j = 0; j < SEQ; ++j) p[j] = (_Float16)((float)p[j] * inv);
    for (int j = SEQ; j < NP; ++j) p[j] = (_Float16)0.f;  // zero padded keys
  }
  __syncthreads();

  // ---- Stage 3: O = P @ V; 24 (head,mt) pairs over 8 waves, both nt each ----
  for (int pair = wave; pair < NH * 4; pair += 8) {
    int h = pair >> 2, mt = pair & 3;
    const _Float16* pb = sp_s + h * NP * NP;
    const _Float16* vb = qkv_s + VOFF + h * HD * NP;  // v^T [d][row]
    v16h a0 = frag_a_lds(pb + (mt * 16 + lrow) * NP, 0, half);
    v16h a1 = frag_a_lds(pb + (mt * 16 + lrow) * NP, 32, half);
#pragma unroll
    for (int nt = 0; nt < 2; ++nt) {
      v8f acc = {};
      v16h b0 = frag_b_f16(vb + (nt * 16 + lrow) * NP, 0, half);
      acc = __builtin_amdgcn_wmma_f32_16x16x32_f16(false, a0, false, b0,
                                                   (short)0, acc, false, false);
      v16h b1 = frag_b_f16(vb + (nt * 16 + lrow) * NP, 32, half);
      acc = __builtin_amdgcn_wmma_f32_16x16x32_f16(false, a1, false, b1,
                                                   (short)0, acc, false, false);
#pragma unroll
      for (int r = 0; r < 8; ++r)
        ao_s[(mt * 16 + half * 8 + r) * DIM + h * HD + nt * 16 + lrow] =
            (_Float16)acc[r];
    }
  }
  __syncthreads();

  // ---- Stage 4: out = AO @ proj_w^T + proj_b (f32 stores, rows < 49) ----
  float* ow = out + win * (SEQ * DIM);
#pragma unroll
  for (int mt = 0; mt < 4; ++mt) {
    v16h a[6];
#pragma unroll
    for (int kk = 0; kk < 6; ++kk)
      a[kk] = frag_a_lds(ao_s + (mt * 16 + lrow) * DIM, kk * 32, half);
    for (int nt = wave; nt < 12; nt += 8) {
      v8f acc = {};
#pragma unroll
      for (int kk = 0; kk < 6; ++kk) {
        v16h b = F16W ? frag_b_f16(wp_h + (size_t)(nt * 16 + lrow) * DIM, kk * 32, half)
                      : frag_b_gw(proj_w, nt * 16 + lrow, DIM, kk * 32, half);
        acc = __builtin_amdgcn_wmma_f32_16x16x32_f16(false, a[kk], false, b,
                                                     (short)0, acc, false, false);
      }
      float bias = proj_b[nt * 16 + lrow];
      float* orow = ow + (size_t)(mt * 16 + half * 8) * DIM + nt * 16 + lrow;
      if (mt < 3) {
        // rows 0..47: always in range, unpredicated stores
#pragma unroll
        for (int r = 0; r < 8; ++r) orow[r * DIM] = acc[r] + bias;
      } else {
#pragma unroll
        for (int r = 0; r < 8; ++r)
          if (mt * 16 + half * 8 + r < SEQ) orow[r * DIM] = acc[r] + bias;
      }
    }
  }
}

extern "C" void kernel_launch(void* const* d_in, const int* in_sizes, int n_in,
                              void* d_out, int out_size, void* d_ws, size_t ws_size,
                              hipStream_t stream) {
  (void)in_sizes; (void)n_in; (void)out_size;
  const float* x      = (const float*)d_in[0];
  const float* qkv_w  = (const float*)d_in[1];
  const float* qkv_b  = (const float*)d_in[2];
  const float* proj_w = (const float*)d_in[3];
  const float* proj_b = (const float*)d_in[4];
  float* out = (float*)d_out;

  const bool use_f16w = (d_ws != nullptr) && (ws_size >= (size_t)WS_BYTES);
  if (use_f16w) {
    _Float16* wq = (_Float16*)d_ws;
    _Float16* wp = wq + QKV_ELEMS;
    hipLaunchKernelGGL(convert_weights_kernel, dim3((QKV_ELEMS + 255) / 256),
                       dim3(256), 0, stream, qkv_w, proj_w, wq, wp);
    hipLaunchKernelGGL((win_attn_kernel<true>), dim3(2048), dim3(256), 0, stream,
                       x, qkv_w, qkv_b, proj_w, proj_b, wq, wp, out);
  } else {
    hipLaunchKernelGGL((win_attn_kernel<false>), dim3(2048), dim3(256), 0, stream,
                       x, qkv_w, qkv_b, proj_w, proj_b,
                       (const _Float16*)nullptr, (const _Float16*)nullptr, out);
  }
}
